// AttentionSlice_23175643529668
// MI455X (gfx1250) — matrix operand order repaired and checked by reading, not compiled, not run on hardware
//
#include <hip/hip_runtime.h>
#include <hip/hip_bf16.h>
#include <stdint.h>

#define NTOK 4096   // H*W
#define CI   64
#define CBIG 128
#define NB   4

typedef _Float16 h16v __attribute__((ext_vector_type(16)));
typedef float    f32x8 __attribute__((ext_vector_type(8)));

union F16Frag { h16v v; uint32_t u[8]; };
union H8 { uint4 q; _Float16 h[8]; };

__device__ __forceinline__ f32x8 wmma_f16(h16v a, h16v b, f32x8 c) {
  // D = A(16x32 f16) * B(32x16 f16) + C(16x16 f32)
  return __builtin_amdgcn_wmma_f32_16x16x32_f16(false, a, false, b, (short)0, c, false, false);
}

// A-fragment K offset within a 32-wide K step (contiguous order: k = 16h + 2j, matching B)
__device__ __forceinline__ int a_koff(int j, int h) {
  return 16 * h + 2 * j;
}

// Async 16B global->LDS copy (GLOBAL_LOAD_ASYNC_TO_LDS_B128, tracked by ASYNCcnt)
__device__ __forceinline__ void async_copy16(const void* gptr, const void* lptr) {
  uint32_t lds = (uint32_t)(uintptr_t)lptr;       // low 32b of LDS-aperture addr = LDS offset
  uint64_t ga  = (uint64_t)(uintptr_t)gptr;
  asm volatile("global_load_async_to_lds_b128 %0, %1, off" :: "v"(lds), "v"(ga) : "memory");
}
__device__ __forceinline__ void async_wait0() {
  asm volatile("s_wait_asynccnt 0" ::: "memory");
}

// ---------------------------------------------------------------------------
// Fold BN (inference) + bias into conv weight: W'[o][c] = W[o][c]*s[o],
// b'[o] = b[o]*s[o] + beta[o] - mean[o]*s[o], s = gamma*rsqrt(var+eps)
// ---------------------------------------------------------------------------
__global__ void fold_kernel(const float* __restrict__ w, const float* __restrict__ b,
                            const float* __restrict__ g, const float* __restrict__ beta,
                            const float* __restrict__ m, const float* __restrict__ v,
                            int cout, int cin, _Float16* __restrict__ wf,
                            float* __restrict__ bf) {
  int i = blockIdx.x * blockDim.x + threadIdx.x;
  if (i < cout * cin) {
    int o = i / cin;
    float s = g[o] * rsqrtf(v[o] + 1e-5f);
    wf[i] = (_Float16)(w[i] * s);
  }
  if (i < cout) {
    float s = g[i] * rsqrtf(v[i] + 1e-5f);
    bf[i] = b[i] * s + beta[i] - m[i] * s;
  }
}

// ---------------------------------------------------------------------------
// QKV projection: dst[b][n][o] = sum_k Wf[o][k] * x[b][k][n] + bias[o], f16 out
// One block = 128 rows (n) x 64 cols (o). 8 waves, each wave a 16x64 slab.
// ---------------------------------------------------------------------------
#define WQ_LD 136   // padded halves per weight row (272B, 16B aligned, conflict-free)
__global__ __launch_bounds__(256)
void qkv_kernel(const float* __restrict__ x, const _Float16* __restrict__ wf,
                const float* __restrict__ bf, _Float16* __restrict__ dst) {
  __shared__ __align__(16) _Float16 Xt[128 * 136];   // Xt[n][k], ld=136
  __shared__ __align__(16) _Float16 Wl[CI * WQ_LD];  // Wl[o][k], ld=136
  const int tid = threadIdx.x;
  const int lane = tid & 31, wv = tid >> 5;
  const int h = lane >> 4, col = lane & 15;
  const int n0 = blockIdx.x * 128;
  const int b  = blockIdx.y;
  const float* xb = x + (size_t)b * CBIG * NTOK;

  // async-stage weights: 64 rows x 16 chunks of 8 halves
#pragma unroll
  for (int r = 0; r < 4; ++r) {
    int q = tid + 256 * r;
    int o = q >> 4, c0 = (q & 15) * 8;
    async_copy16(&wf[o * CBIG + c0], &Wl[o * WQ_LD + c0]);
  }
  // transpose-stage activations (f32 -> f16)
  for (int i = tid; i < 128 * 128; i += 256) {
    int n = i & 127, k = i >> 7;
    Xt[n * 136 + k] = (_Float16)xb[(size_t)k * NTOK + n0 + n];
  }
  async_wait0();
  __syncthreads();

  const int row = 16 * wv + col;  // local row for this wave's A fragment
  F16Frag afr[4];
#pragma unroll
  for (int ks = 0; ks < 4; ++ks)
#pragma unroll
    for (int j = 0; j < 8; ++j)
      afr[ks].u[j] = *(const uint32_t*)&Xt[row * 136 + 32 * ks + a_koff(j, h)];

#pragma unroll
  for (int ct = 0; ct < 4; ++ct) {
    const int o = 16 * ct + col;
    f32x8 acc = {};
#pragma unroll
    for (int ks = 0; ks < 4; ++ks) {
      F16Frag bfr;
#pragma unroll
      for (int j = 0; j < 8; ++j)
        bfr.u[j] = *(const uint32_t*)&Wl[o * WQ_LD + 32 * ks + 16 * h + 2 * j];
      acc = wmma_f16(afr[ks].v, bfr.v, acc);
    }
    const float bb = bf[o];
#pragma unroll
    for (int vv = 0; vv < 8; ++vv) {
      int n = n0 + 16 * wv + vv + 8 * h;
      dst[((size_t)b * NTOK + n) * CI + o] = (_Float16)(acc[vv] + bb);
    }
  }
}

// ---------------------------------------------------------------------------
// Flash attention: O[n][c] = softmax_m(Q[n]·K[m]) · V[m][c], d = 64, N = 4096
// Block = 128 query rows; wave = 16 rows. Loop over 64-wide key tiles.
// ---------------------------------------------------------------------------
#define KT_LD 72    // padded halves per K row (144B, 16B aligned, conflict-free)
__global__ __launch_bounds__(256)
void flash_kernel(const _Float16* __restrict__ Qm, const _Float16* __restrict__ Km,
                  const _Float16* __restrict__ Vm, _Float16* __restrict__ Og) {
  __shared__ __align__(16) _Float16 Kt[64 * KT_LD];  // Kt[m][c]
  __shared__ __align__(16) _Float16 VtT[64 * 68];    // VtT[c][m]
  __shared__ __align__(16) _Float16 Pt[8 * 16 * 68]; // per-wave 16x64 P tile
  const int tid = threadIdx.x, lane = tid & 31, wv = tid >> 5;
  const int h = lane >> 4, col = lane & 15;
  const int b = blockIdx.y, n0 = blockIdx.x * 128;
  const size_t base = (size_t)b * NTOK * CI;

  // persistent Q fragments (16 rows x 64 ch -> 2 K-steps)
  const int qrow = n0 + 16 * wv + col;
  F16Frag qf[2];
#pragma unroll
  for (int ks = 0; ks < 2; ++ks)
#pragma unroll
    for (int j = 0; j < 8; ++j)
      qf[ks].u[j] = *(const uint32_t*)&Qm[base + (size_t)qrow * CI + 32 * ks + a_koff(j, h)];

  f32x8 o_acc[4];
#pragma unroll
  for (int t = 0; t < 4; ++t) { f32x8 z = {}; o_acc[t] = z; }
  float mi[8], li[8];
#pragma unroll
  for (int vv = 0; vv < 8; ++vv) { mi[vv] = -1e30f; li[vv] = 0.0f; }
  _Float16* Pw = &Pt[wv * 16 * 68];

  for (int mt = 0; mt < NTOK / 64; ++mt) {
    const int m0 = mt * 64;
    __syncthreads();  // previous iteration done with Kt/VtT
    // K tile: async DMA straight into LDS (64 rows x 8 chunks of 16B)
#pragma unroll
    for (int r = 0; r < 2; ++r) {
      int q = tid + 256 * r;
      int m = q >> 3, c0 = (q & 7) * 8;
      async_copy16(&Km[base + (size_t)(m0 + m) * CI + c0], &Kt[m * KT_LD + c0]);
    }
    // V tile: vectorized load + transpose scatter (needs [c][m] layout)
#pragma unroll
    for (int r = 0; r < 2; ++r) {
      int q = tid + 256 * r;
      int m = q >> 3, c0 = (q & 7) * 8;
      H8 vv8;
      vv8.q = *(const uint4*)&Vm[base + (size_t)(m0 + m) * CI + c0];
#pragma unroll
      for (int e = 0; e < 8; ++e) VtT[(c0 + e) * 68 + m] = vv8.h[e];
    }
    async_wait0();
    __syncthreads();

    // S = Q K^T  (4 col tiles of 16, K-dim = 64 channels -> 2 steps)
    f32x8 s[4];
#pragma unroll
    for (int ct = 0; ct < 4; ++ct) {
      f32x8 acc = {};
      const int m = 16 * ct + col;
#pragma unroll
      for (int ks = 0; ks < 2; ++ks) {
        F16Frag bfr;
#pragma unroll
        for (int j = 0; j < 8; ++j)
          bfr.u[j] = *(const uint32_t*)&Kt[m * KT_LD + 32 * ks + 16 * h + 2 * j];
        acc = wmma_f16(qf[ks].v, bfr.v, acc);
      }
      s[ct] = acc;
    }

    // online softmax; lane's VGPR vv holds row (vv + 8h), col (L&15)
    float scale[8];
#pragma unroll
    for (int vv = 0; vv < 8; ++vv) {
      float mx = fmaxf(fmaxf(s[0][vv], s[1][vv]), fmaxf(s[2][vv], s[3][vv]));
#pragma unroll
      for (int off = 1; off < 16; off <<= 1) mx = fmaxf(mx, __shfl_xor(mx, off, 16));
      const float mnew = fmaxf(mi[vv], mx);
      scale[vv] = __expf(mi[vv] - mnew);
      float rs = 0.0f;
#pragma unroll
      for (int t = 0; t < 4; ++t) {
        float p = __expf(s[t][vv] - mnew);
        s[t][vv] = p;
        rs += p;
      }
#pragma unroll
      for (int off = 1; off < 16; off <<= 1) rs += __shfl_xor(rs, off, 16);
      li[vv] = li[vv] * scale[vv] + rs;
      mi[vv] = mnew;
    }

    // rescale accumulator; spill P (f16) to per-wave LDS tile
#pragma unroll
    for (int t = 0; t < 4; ++t)
#pragma unroll
      for (int vv = 0; vv < 8; ++vv) {
        o_acc[t][vv] *= scale[vv];
        Pw[(vv + 8 * h) * 68 + 16 * t + col] = (_Float16)s[t][vv];
      }
    __syncthreads();  // make P visible across lanes (and order DS ops)

    // O += P @ V  (K-dim = 64 keys -> 2 steps)
    F16Frag pf[2];
#pragma unroll
    for (int ks = 0; ks < 2; ++ks)
#pragma unroll
      for (int j = 0; j < 8; ++j)
        pf[ks].u[j] = *(const uint32_t*)&Pw[col * 68 + 32 * ks + a_koff(j, h)];
#pragma unroll
    for (int ct = 0; ct < 4; ++ct)
#pragma unroll
      for (int ks = 0; ks < 2; ++ks) {
        F16Frag bfr;
#pragma unroll
        for (int j = 0; j < 8; ++j)
          bfr.u[j] = *(const uint32_t*)&VtT[(16 * ct + col) * 68 + 32 * ks + 16 * h + 2 * j];
        o_acc[ct] = wmma_f16(pf[ks].v, bfr.v, o_acc[ct]);
      }
  }

  // normalize and store O as f16 [b][n][c]
#pragma unroll
  for (int ct = 0; ct < 4; ++ct)
#pragma unroll
    for (int vv = 0; vv < 8; ++vv) {
      int n = n0 + 16 * wv + vv + 8 * h;
      float val = o_acc[ct][vv] / li[vv];
      Og[base + (size_t)n * CI + 16 * ct + col] = (_Float16)val;
    }
}

// ---------------------------------------------------------------------------
// Output projection: out[b][o][n] = sum_c Wo[o][c]*O[b][n][c] + bo[o], o<128
// D tiles staged through LDS so global stores are coalesced float4.
// ---------------------------------------------------------------------------
#define WO_LD 72    // padded halves per Wout row
#define OT_LD 132   // padded floats per output-tile row
__global__ __launch_bounds__(256)
void outproj_kernel(const _Float16* __restrict__ Og, const _Float16* __restrict__ Wo,
                    const float* __restrict__ bo, float* __restrict__ out) {
  __shared__ __align__(16) _Float16 Wl[CBIG * WO_LD];  // Wl[o][c]
  __shared__ __align__(16) float Ot[CBIG * OT_LD];     // Ot[o][n_local]
  const int tid = threadIdx.x, lane = tid & 31, wv = tid >> 5;
  const int h = lane >> 4, col = lane & 15;
  const int b = blockIdx.y, n0 = blockIdx.x * 128;

  // async-stage Wout: 128 rows x 8 chunks of 16B
#pragma unroll
  for (int r = 0; r < 4; ++r) {
    int q = tid + 256 * r;
    int o = q >> 3, c0 = (q & 7) * 8;
    async_copy16(&Wo[o * CI + c0], &Wl[o * WO_LD + c0]);
  }

  const size_t base = (size_t)b * NTOK * CI;
  const int row = n0 + 16 * wv + col;
  F16Frag af[2];
#pragma unroll
  for (int ks = 0; ks < 2; ++ks)
#pragma unroll
    for (int j = 0; j < 8; ++j)
      af[ks].u[j] = *(const uint32_t*)&Og[base + (size_t)row * CI + 32 * ks + a_koff(j, h)];
  async_wait0();
  __syncthreads();

#pragma unroll
  for (int ot = 0; ot < 8; ++ot) {
    const int o = 16 * ot + col;
    f32x8 acc = {};
#pragma unroll
    for (int ks = 0; ks < 2; ++ks) {
      F16Frag bfr;
#pragma unroll
      for (int j = 0; j < 8; ++j)
        bfr.u[j] = *(const uint32_t*)&Wl[o * WO_LD + 32 * ks + 16 * h + 2 * j];
      acc = wmma_f16(af[ks].v, bfr.v, acc);
    }
    const float bb = bo[o];
#pragma unroll
    for (int vv = 0; vv < 8; ++vv)
      Ot[o * OT_LD + 16 * wv + vv + 8 * h] = acc[vv] + bb;
  }
  __syncthreads();

  // cooperative coalesced store: 128 (o) x 128 (n) floats as float4
  for (int i = tid; i < CBIG * 32; i += 256) {
    int o = i >> 5, n4 = (i & 31) * 4;
    float4 vv4 = *(const float4*)&Ot[o * OT_LD + n4];
    *(float4*)&out[((size_t)b * 256 + o) * NTOK + n0 + n4] = vv4;
  }
}

// Concat: out channels [128,256) = x1
__global__ void concat_kernel(const float* __restrict__ x1, float* __restrict__ out) {
  size_t i = (size_t)blockIdx.x * blockDim.x + threadIdx.x;  // float4 units
  const size_t perb = (size_t)CBIG * NTOK / 4;
  size_t b = i / perb, r = i % perb;
  if (b < NB) {
    const float4* src = (const float4*)(x1 + b * (size_t)CBIG * NTOK);
    float4* dst = (float4*)(out + ((b * 256 + 128) * (size_t)NTOK));
    dst[r] = src[r];
  }
}

extern "C" void kernel_launch(void* const* d_in, const int* in_sizes, int n_in,
                              void* d_out, int out_size, void* d_ws, size_t ws_size,
                              hipStream_t stream) {
  (void)in_sizes; (void)n_in; (void)out_size; (void)ws_size;
  const float* x1 = (const float*)d_in[0];
  const float* x2 = (const float*)d_in[1];
  // per-layer params start: theta@2, phi@8, g@14, wout@20 (w,b,gamma,beta,mean,var)
  const float** P = (const float**)d_in;

  char* ws = (char*)d_ws;
  size_t off = 0;
  auto alloc = [&](size_t bytes) -> void* {
    void* p = ws + off;
    off = (off + bytes + 255) & ~(size_t)255;
    return p;
  };
  _Float16* Wth = (_Float16*)alloc(CI * CBIG * 2);
  _Float16* Wph = (_Float16*)alloc(CI * CBIG * 2);
  _Float16* Wg  = (_Float16*)alloc(CI * CBIG * 2);
  _Float16* Wo  = (_Float16*)alloc(CBIG * CI * 2);
  float* bth = (float*)alloc(CI * 4);
  float* bph = (float*)alloc(CI * 4);
  float* bg  = (float*)alloc(CI * 4);
  float* bo  = (float*)alloc(CBIG * 4);
  const size_t qkvBytes = (size_t)NB * NTOK * CI * 2;
  _Float16* Q  = (_Float16*)alloc(qkvBytes);
  _Float16* K  = (_Float16*)alloc(qkvBytes);
  _Float16* V  = (_Float16*)alloc(qkvBytes);
  _Float16* Og = (_Float16*)alloc(qkvBytes);

  const int foldBlocks = (CBIG * CI + 255) / 256;
  fold_kernel<<<foldBlocks, 256, 0, stream>>>(P[2],  P[3],  P[4],  P[5],  P[6],  P[7],  CI,   CBIG, Wth, bth);
  fold_kernel<<<foldBlocks, 256, 0, stream>>>(P[8],  P[9],  P[10], P[11], P[12], P[13], CI,   CBIG, Wph, bph);
  fold_kernel<<<foldBlocks, 256, 0, stream>>>(P[14], P[15], P[16], P[17], P[18], P[19], CI,   CBIG, Wg,  bg);
  fold_kernel<<<foldBlocks, 256, 0, stream>>>(P[20], P[21], P[22], P[23], P[24], P[25], CBIG, CI,   Wo,  bo);

  dim3 grid(NTOK / 128, NB);
  qkv_kernel<<<grid, 256, 0, stream>>>(x1, Wth, bth, Q);
  qkv_kernel<<<grid, 256, 0, stream>>>(x2, Wph, bph, K);
  qkv_kernel<<<grid, 256, 0, stream>>>(x2, Wg,  bg,  V);
  flash_kernel<<<grid, 256, 0, stream>>>(Q, K, V, Og);
  outproj_kernel<<<grid, 256, 0, stream>>>(Og, Wo, bo, (float*)d_out);
  concat_kernel<<<(NB * CBIG * NTOK / 4 + 255) / 256, 256, 0, stream>>>(x1, (float*)d_out);
}
